// MPNN_50010599194664
// MI455X (gfx1250) — compile-verified, matrix operand order checked by reading, not run on hardware
//
#include <hip/hip_runtime.h>
#include <hip/hip_bf16.h>
#include <math.h>

// ---------------------------------------------------------------------------
// Types for WMMA (gfx1250, wave32)
// ---------------------------------------------------------------------------
typedef _Float16 v16h __attribute__((ext_vector_type(16)));
typedef _Float16 v8h  __attribute__((ext_vector_type(8)));
typedef float    v8f  __attribute__((ext_vector_type(8)));

// Problem constants (match reference)
#define N_ATOMS   100000
#define ATOM_FDIM 133
#define HIDDEN    512
#define MAX_NEI   6
#define N_MOLS    4096
#define N_PAIRS   8192
#define MP        100096   // N_ATOMS padded to multiple of 128 (block tile M)
#define KP1       160      // 133 padded to multiple of 32
#define KO        672      // 645 padded to multiple of 32
#define FF        2048
#define FF1       1024

// ---------------------------------------------------------------------------
// f32 -> f16 convert with zero padding (row-major A operands).
// ---------------------------------------------------------------------------
__global__ void convert_pad_kernel(const float* __restrict__ src, _Float16* __restrict__ dst,
                                   int srcRows, int srcCols, int srcLd,
                                   int dstRows, int dstLd, int dstColOff, int regionCols) {
    long long total = (long long)dstRows * (long long)regionCols;
    for (long long idx = (long long)blockIdx.x * blockDim.x + threadIdx.x; idx < total;
         idx += (long long)gridDim.x * blockDim.x) {
        int r = (int)(idx / regionCols);
        int c = (int)(idx - (long long)r * regionCols);
        float v = 0.0f;
        if (r < srcRows && c < srcCols) v = src[(size_t)r * srcLd + c];
        dst[(size_t)r * dstLd + dstColOff + c] = (_Float16)v;
    }
}

// ---------------------------------------------------------------------------
// Weight conversion: f32 W[K,N] (row-major) -> f16 WT[N,KP] (K-major, padded).
// Done once per launch so the GEMM can stage B with contiguous 16B copies.
// ---------------------------------------------------------------------------
__global__ void convert_pad_t_kernel(const float* __restrict__ src, _Float16* __restrict__ dst,
                                     int srcK, int srcN, int srcLd, int dstN, int dstKp) {
    long long total = (long long)dstN * (long long)dstKp;
    for (long long idx = (long long)blockIdx.x * blockDim.x + threadIdx.x; idx < total;
         idx += (long long)gridDim.x * blockDim.x) {
        int n = (int)(idx / dstKp);
        int k = (int)(idx - (long long)n * dstKp);
        float v = 0.0f;
        if (k < srcK && n < srcN) v = src[(size_t)k * srcLd + n];
        dst[(size_t)n * dstKp + k] = (_Float16)v;
    }
}

// ---------------------------------------------------------------------------
// Neighbor gather + sum (optionally ReLU on read). f16 output into region.
// ---------------------------------------------------------------------------
template<bool RELU_SRC>
__global__ void gather_sum_kernel(const float* __restrict__ src, const int* __restrict__ nei,
                                  _Float16* __restrict__ dst, int dstLd, int colOff,
                                  int nAtoms) {
    int r = blockIdx.x;
    for (int h = threadIdx.x; h < HIDDEN; h += blockDim.x) {
        float s = 0.0f;
        if (r < nAtoms) {
#pragma unroll
            for (int i = 0; i < MAX_NEI; ++i) {
                int a = nei[r * MAX_NEI + i];
                float v = src[(size_t)a * HIDDEN + h];
                if (RELU_SRC) v = fmaxf(v, 0.0f);
                s += v;
            }
        }
        dst[(size_t)r * dstLd + colOff + h] = (_Float16)s;
    }
}

// ---------------------------------------------------------------------------
// WMMA 16-bit fragment from LDS tile with 32-half rows (two 16B ds loads).
// Lane L (<16 -> row=L, base=0 ; >=16 -> row=L-16, K-base=+8):
//   elems 0..7  : K = base .. base+7
//   elems 8..15 : K = base+16 .. base+23
// ---------------------------------------------------------------------------
__device__ __forceinline__ v16h load_frag16(const _Float16* rowPtr, int halfBase) {
    v8h lo = *reinterpret_cast<const v8h*>(rowPtr + halfBase);
    v8h hi = *reinterpret_cast<const v8h*>(rowPtr + halfBase + 16);
    return __builtin_shufflevector(lo, hi, 0, 1, 2, 3, 4, 5, 6, 7,
                                           8, 9, 10, 11, 12, 13, 14, 15);
}

// ---------------------------------------------------------------------------
// Tiled f16 WMMA GEMM, f32 accumulate, double-buffered LDS.
//   C[M,N] = epilogue( A16[M,K] @ BT16[N,K]^T )
//   Block tile 128x64 (256 threads = 8 waves), wave tile 32x32:
//   2 A-frags + 2 B-frags -> 4 v_wmma per K-step (8 ds_load_b128 : 4 wmma).
//   B is pre-transposed (K-major), so BOTH tiles stage with ds_store_b128.
//   One barrier per K-step; next K-tile global loads issued before the WMMAs.
// M % 128 == 0, N % 64 == 0, K % 32 == 0 guaranteed by padding.
// ---------------------------------------------------------------------------
template<bool RELU, bool BIAS, bool ADDMAT, bool OUTF16>
__global__ __launch_bounds__(256) void gemm_wmma_kernel(
    const _Float16* __restrict__ A, const _Float16* __restrict__ BT,
    void* __restrict__ Cv, const float* __restrict__ bias,
    const float* __restrict__ add, int M, int N, int K) {
    __shared__ __align__(16) _Float16 sA[2][128 * 32];   // [m][k]
    __shared__ __align__(16) _Float16 sBT[2][64 * 32];   // [n][k]

    const int tid  = threadIdx.x;
    const int lane = tid & 31;
    const int wave = tid >> 5;
    const int mBase = blockIdx.y * 128;
    const int nBase = blockIdx.x * 64;

    const int mOff = (wave & 3) * 32;   // wave's M sub-tile (32 rows)
    const int nOff = (wave >> 2) * 32;  // wave's N sub-tile (32 cols)

    v8f acc00 = {0.f, 0.f, 0.f, 0.f, 0.f, 0.f, 0.f, 0.f};
    v8f acc01 = acc00, acc10 = acc00, acc11 = acc00;

    // cooperative-load coords: A 128x32 (16 halfs/thread), BT 64x32 (8/thread)
    const int ar = tid >> 1;          // 0..127
    const int ac = (tid & 1) * 16;    // 0 or 16
    const int bn = tid >> 2;          // 0..63
    const int bc = (tid & 3) * 8;     // 0,8,16,24

    const int halfBase = (lane >> 4) << 3;     // 0 or 8
    const int fragRow  = lane & 15;

    const _Float16* aPtr = A  + (size_t)(mBase + ar) * K + ac;
    const _Float16* bPtr = BT + (size_t)(nBase + bn) * K + bc;

    // ---- prologue: stage K-tile 0 into buffer 0 ----
    {
        v8h a0 = *reinterpret_cast<const v8h*>(aPtr);
        v8h a1 = *reinterpret_cast<const v8h*>(aPtr + 8);
        v8h b0 = *reinterpret_cast<const v8h*>(bPtr);
        *reinterpret_cast<v8h*>(&sA[0][ar * 32 + ac])     = a0;
        *reinterpret_cast<v8h*>(&sA[0][ar * 32 + ac + 8]) = a1;
        *reinterpret_cast<v8h*>(&sBT[0][bn * 32 + bc])    = b0;
    }

    int cur = 0;
    for (int k0 = 0; k0 < K; k0 += 32) {
        __syncthreads();
        const bool hasNext = (k0 + 32) < K;

        // issue next K-tile global loads early (overlap with WMMA)
        v8h na0, na1, nb;
        if (hasNext) {
            na0 = *reinterpret_cast<const v8h*>(aPtr + 32);
            na1 = *reinterpret_cast<const v8h*>(aPtr + 40);
            nb  = *reinterpret_cast<const v8h*>(bPtr + 32);
        }
        // speculative prefetch of the K-tile after that (safe: dropped on fault)
        __builtin_prefetch(aPtr + 64, 0, 0);
        __builtin_prefetch(bPtr + 64, 0, 0);

        // ---- fragments + 4 WMMAs ----
        const _Float16* bufA = sA[cur];
        const _Float16* bufB = sBT[cur];
        v16h afrag0 = load_frag16(&bufA[(mOff + fragRow)      * 32], halfBase);
        v16h afrag1 = load_frag16(&bufA[(mOff + 16 + fragRow) * 32], halfBase);
        v16h bfrag0 = load_frag16(&bufB[(nOff + fragRow)      * 32], halfBase);
        v16h bfrag1 = load_frag16(&bufB[(nOff + 16 + fragRow) * 32], halfBase);

        acc00 = __builtin_amdgcn_wmma_f32_16x16x32_f16(false, afrag0, false, bfrag0,
                                                       (short)0, acc00, false, false);
        acc01 = __builtin_amdgcn_wmma_f32_16x16x32_f16(false, afrag0, false, bfrag1,
                                                       (short)0, acc01, false, false);
        acc10 = __builtin_amdgcn_wmma_f32_16x16x32_f16(false, afrag1, false, bfrag0,
                                                       (short)0, acc10, false, false);
        acc11 = __builtin_amdgcn_wmma_f32_16x16x32_f16(false, afrag1, false, bfrag1,
                                                       (short)0, acc11, false, false);

        // ---- stage next K-tile into the other buffer (all b128 stores) ----
        if (hasNext) {
            _Float16* dA = sA[cur ^ 1];
            _Float16* dB = sBT[cur ^ 1];
            *reinterpret_cast<v8h*>(&dA[ar * 32 + ac])     = na0;
            *reinterpret_cast<v8h*>(&dA[ar * 32 + ac + 8]) = na1;
            *reinterpret_cast<v8h*>(&dB[bn * 32 + bc])     = nb;
        }
        aPtr += 32;
        bPtr += 32;
        cur ^= 1;
    }

    // ---- epilogue: C/D layout: lane<16 -> M=v, lane>=16 -> M=v+8; N=lane%16 ----
    const int rBase = mBase + mOff + ((lane >> 4) << 3);
    const int c0    = nBase + nOff + (lane & 15);

    auto store_tile = [&](const v8f& acc, int row0, int col) {
        const float bval = BIAS ? bias[col] : 0.0f;
        const float* ap = ADDMAT ? (add + (size_t)row0 * N + col) : nullptr;
        _Float16* cph = OUTF16 ? (reinterpret_cast<_Float16*>(Cv) + (size_t)row0 * N + col) : nullptr;
        float*    cpf = OUTF16 ? nullptr : (reinterpret_cast<float*>(Cv) + (size_t)row0 * N + col);
#pragma unroll
        for (int v = 0; v < 8; ++v) {
            float val = acc[v];
            if (BIAS)   val += bval;
            if (ADDMAT) { val += *ap; ap += N; }
            if (RELU)   val = fmaxf(val, 0.0f);
            if (OUTF16) { *cph = (_Float16)val; cph += N; }
            else        { *cpf = val;           cpf += N; }
        }
    };
    store_tile(acc00, rBase,      c0);
    store_tile(acc01, rBase,      c0 + 16);
    store_tile(acc10, rBase + 16, c0);
    store_tile(acc11, rBase + 16, c0 + 16);
}

// ---------------------------------------------------------------------------
// Segment mean over sorted mol_ids (binary search for bounds).
// ---------------------------------------------------------------------------
__global__ void segment_mean_kernel(const float* __restrict__ ah, const int* __restrict__ mol_ids,
                                    float* __restrict__ mv, int nAtoms) {
    int m = blockIdx.x;
    __shared__ int sLo, sHi;
    if (threadIdx.x == 0) {
        int lo = 0, hi = nAtoms;
        while (lo < hi) { int mid = (lo + hi) >> 1; if (mol_ids[mid] < m) lo = mid + 1; else hi = mid; }
        sLo = lo;
        hi = nAtoms;
        while (lo < hi) { int mid = (lo + hi) >> 1; if (mol_ids[mid] <= m) lo = mid + 1; else hi = mid; }
        sHi = lo;
    }
    __syncthreads();
    int lo = sLo, hi = sHi;
    float inv = 1.0f / fmaxf((float)(hi - lo), 1.0f);
    for (int h = threadIdx.x; h < HIDDEN; h += blockDim.x) {
        float s = 0.0f;
        for (int a = lo; a < hi; ++a) s += ah[(size_t)a * HIDDEN + h];
        mv[(size_t)m * HIDDEN + h] = s * inv;
    }
}

// ---------------------------------------------------------------------------
// Pairwise feature: [e1+e2, e1*e2, e1, e2] -> f16.
// ---------------------------------------------------------------------------
__global__ void pair_feature_kernel(const float* __restrict__ mv, const int* __restrict__ edges,
                                    _Float16* __restrict__ feat) {
    int p = blockIdx.x;
    int i = edges[p];
    int j = edges[N_PAIRS + p];
    _Float16* f = feat + (size_t)p * (4 * HIDDEN);
    for (int h = threadIdx.x; h < HIDDEN; h += blockDim.x) {
        float e1 = mv[(size_t)i * HIDDEN + h];
        float e2 = mv[(size_t)j * HIDDEN + h];
        f[h]               = (_Float16)(e1 + e2);
        f[HIDDEN + h]      = (_Float16)(e1 * e2);
        f[2 * HIDDEN + h]  = (_Float16)e1;
        f[3 * HIDDEN + h]  = (_Float16)e2;
    }
}

// ---------------------------------------------------------------------------
// Final 1024-dot + bias + sigmoid.
// ---------------------------------------------------------------------------
__global__ void final_dot_kernel(const _Float16* __restrict__ h2, const float* __restrict__ w,
                                 const float* __restrict__ b, float* __restrict__ out) {
    int p = blockIdx.x;
    __shared__ float red[256];
    float s = 0.0f;
    for (int k = threadIdx.x; k < FF1; k += 256)
        s += (float)h2[(size_t)p * FF1 + k] * w[k];
    red[threadIdx.x] = s;
    __syncthreads();
    for (int off = 128; off > 0; off >>= 1) {
        if (threadIdx.x < off) red[threadIdx.x] += red[threadIdx.x + off];
        __syncthreads();
    }
    if (threadIdx.x == 0) {
        float x = red[0] + b[0];
        out[p] = 1.0f / (1.0f + __expf(-x));
    }
}

// ---------------------------------------------------------------------------
// Host orchestration
// ---------------------------------------------------------------------------
extern "C" void kernel_launch(void* const* d_in, const int* in_sizes, int n_in,
                              void* d_out, int out_size, void* d_ws, size_t ws_size,
                              hipStream_t stream) {
    (void)in_sizes; (void)n_in; (void)out_size; (void)ws_size;

    const float* f_atoms     = (const float*)d_in[0];
    const int*   a_neighbors = (const int*)  d_in[1];
    const int*   mol_ids     = (const int*)  d_in[2];
    const int*   batch_edges = (const int*)  d_in[3];
    const float* W_i         = (const float*)d_in[4];
    const float* W_h         = (const float*)d_in[5];
    const float* W_o         = (const float*)d_in[6];
    const float* b_o         = (const float*)d_in[7];
    const float* W_ffn_i     = (const float*)d_in[8];
    const float* b_ffn_i     = (const float*)d_in[9];
    const float* W_ffn_1     = (const float*)d_in[10];
    const float* b_ffn_1     = (const float*)d_in[11];
    const float* W_ffn_2     = (const float*)d_in[12];
    const float* b_ffn_2     = (const float*)d_in[13];
    float* out = (float*)d_out;

    // ---- workspace carve-up ----
    char* p = (char*)d_ws;
    auto alloc = [&](size_t bytes) -> void* {
        void* r = (void*)p;
        p += (bytes + 255) & ~(size_t)255;
        return r;
    };
    _Float16* fA16    = (_Float16*)alloc((size_t)MP * KP1 * 2);
    _Float16* WiT16   = (_Float16*)alloc((size_t)HIDDEN * KP1 * 2);   // [N=512, K=160]
    _Float16* WhT16   = (_Float16*)alloc((size_t)HIDDEN * HIDDEN * 2);// [512, 512]
    _Float16* WoT16   = (_Float16*)alloc((size_t)HIDDEN * KO * 2);    // [512, 672]
    _Float16* WffiT16 = (_Float16*)alloc((size_t)FF * FF * 2);        // [2048, 2048]
    _Float16* Wff1T16 = (_Float16*)alloc((size_t)FF1 * FF * 2);       // [1024, 2048]
    float*    inp     = (float*)   alloc((size_t)MP * HIDDEN * 4);
    _Float16* msg16   = (_Float16*)alloc((size_t)MP * HIDDEN * 2);
    float*    message = (float*)   alloc((size_t)MP * HIDDEN * 4);
    _Float16* ain16   = (_Float16*)alloc((size_t)MP * KO * 2);
    float*    ah      = (float*)   alloc((size_t)MP * HIDDEN * 4);
    float*    mv      = (float*)   alloc((size_t)N_MOLS * HIDDEN * 4);
    _Float16* feat16  = (_Float16*)alloc((size_t)N_PAIRS * FF * 2);
    _Float16* h1_16   = (_Float16*)alloc((size_t)N_PAIRS * FF * 2);
    _Float16* h2_16   = (_Float16*)alloc((size_t)N_PAIRS * FF1 * 2);

    const dim3 blk(256);
    auto cvtGrid = [](long long total) {
        long long g = (total + 255) / 256;
        if (g > 16384) g = 16384;
        return dim3((unsigned)g);
    };

    // ---- 1a. A-side conversions (row-major, padded) ----
    convert_pad_kernel<<<cvtGrid((long long)MP * KP1), blk, 0, stream>>>(
        f_atoms, fA16, N_ATOMS, ATOM_FDIM, ATOM_FDIM, MP, KP1, 0, KP1);
    // a_input columns [0,133) = f_atoms ; columns [645,672) = zero pad
    convert_pad_kernel<<<cvtGrid((long long)MP * ATOM_FDIM), blk, 0, stream>>>(
        f_atoms, ain16, N_ATOMS, ATOM_FDIM, ATOM_FDIM, MP, KO, 0, ATOM_FDIM);
    convert_pad_kernel<<<cvtGrid((long long)MP * (KO - (ATOM_FDIM + HIDDEN))), blk, 0, stream>>>(
        f_atoms, ain16, 0, 0, ATOM_FDIM, MP, KO, ATOM_FDIM + HIDDEN, KO - (ATOM_FDIM + HIDDEN));

    // ---- 1b. weight conversions: W[K,N] -> WT[N,KP] (K-major, zero-padded K) ----
    convert_pad_t_kernel<<<cvtGrid((long long)HIDDEN * KP1), blk, 0, stream>>>(
        W_i, WiT16, ATOM_FDIM, HIDDEN, HIDDEN, HIDDEN, KP1);
    convert_pad_t_kernel<<<cvtGrid((long long)HIDDEN * HIDDEN), blk, 0, stream>>>(
        W_h, WhT16, HIDDEN, HIDDEN, HIDDEN, HIDDEN, HIDDEN);
    convert_pad_t_kernel<<<cvtGrid((long long)HIDDEN * KO), blk, 0, stream>>>(
        W_o, WoT16, ATOM_FDIM + HIDDEN, HIDDEN, HIDDEN, HIDDEN, KO);
    convert_pad_t_kernel<<<cvtGrid((long long)FF * FF), blk, 0, stream>>>(
        W_ffn_i, WffiT16, FF, FF, FF, FF, FF);
    convert_pad_t_kernel<<<cvtGrid((long long)FF1 * FF), blk, 0, stream>>>(
        W_ffn_1, Wff1T16, FF, FF1, FF1, FF1, FF);

    // ---- 2. inp = f_atoms @ W_i ----
    gemm_wmma_kernel<false, false, false, false><<<dim3(HIDDEN / 64, MP / 128), blk, 0, stream>>>(
        fA16, WiT16, (void*)inp, nullptr, nullptr, MP, HIDDEN, KP1);

    // ---- 3. first gather: sum relu(inp)[neighbors] -> f16 ----
    gather_sum_kernel<true><<<dim3(MP), blk, 0, stream>>>(
        inp, a_neighbors, msg16, HIDDEN, 0, N_ATOMS);

    // ---- 4. message = relu(inp + msg @ W_h) ----
    gemm_wmma_kernel<true, false, true, false><<<dim3(HIDDEN / 64, MP / 128), blk, 0, stream>>>(
        msg16, WhT16, (void*)message, nullptr, inp, MP, HIDDEN, HIDDEN);

    // ---- 5. second gather -> a_input columns [133,645) ----
    gather_sum_kernel<false><<<dim3(MP), blk, 0, stream>>>(
        message, a_neighbors, ain16, KO, ATOM_FDIM, N_ATOMS);

    // ---- 6. atom_hiddens = relu(a_input @ W_o + b_o) ----
    gemm_wmma_kernel<true, true, false, false><<<dim3(HIDDEN / 64, MP / 128), blk, 0, stream>>>(
        ain16, WoT16, (void*)ah, b_o, nullptr, MP, HIDDEN, KO);

    // ---- 7. segment mean per molecule ----
    segment_mean_kernel<<<dim3(N_MOLS), blk, 0, stream>>>(ah, mol_ids, mv, N_ATOMS);

    // ---- 8. pairwise features -> f16 ----
    pair_feature_kernel<<<dim3(N_PAIRS), blk, 0, stream>>>(mv, batch_edges, feat16);

    // ---- 9. h1 = relu(feature @ W_ffn_i + b_ffn_i), f16 out ----
    gemm_wmma_kernel<true, true, false, true><<<dim3(FF / 64, N_PAIRS / 128), blk, 0, stream>>>(
        feat16, WffiT16, (void*)h1_16, b_ffn_i, nullptr, N_PAIRS, FF, FF);

    // ---- 10. h2 = relu(h1 @ W_ffn_1 + b_ffn_1), f16 out ----
    gemm_wmma_kernel<true, true, false, true><<<dim3(FF1 / 64, N_PAIRS / 128), blk, 0, stream>>>(
        h1_16, Wff1T16, (void*)h2_16, b_ffn_1, nullptr, N_PAIRS, FF1, FF);

    // ---- 11. out = sigmoid(h2 @ W_ffn_2 + b_ffn_2) ----
    final_dot_kernel<<<dim3(N_PAIRS), blk, 0, stream>>>(h2_16, W_ffn_2, b_ffn_2, out);
}
